// SCNDecoder_17188459118868
// MI455X (gfx1250) — matrix operand order repaired and checked by reading, not compiled, not run on hardware
//
#include <hip/hip_runtime.h>
#include <hip/hip_bf16.h>
#include <math.h>

#define N_CRYST 64
#define ATOMS   32
#define NATOM   2048
#define NBRS    20
#define NEDGE   40960
#define CH      128
#define HID     256
#define NB      128
#define NLAYER  8
#define K_SPH   19
#define KC      (K_SPH*CH)   // 2432
#define MAXZ    100
#define ZDIM    256
#define FOUR_PI 12.566370614359172f
#define NTILE   (NEDGE/16)   // 2560 edge tiles

typedef __attribute__((ext_vector_type(16))) __bf16          v16bf;
typedef __attribute__((ext_vector_type(16))) unsigned short  v16us;
typedef __attribute__((ext_vector_type(8)))  float           v8f;

// ---------- helpers ----------

__device__ __forceinline__ unsigned short f2bf(float f) {
  unsigned int u = __float_as_uint(f);
  unsigned int r = u + 0x7FFFu + ((u >> 16) & 1u);   // round-to-nearest-even
  return (unsigned short)(r >> 16);
}

// Fragment load: per-lane contiguous 16 bf16 (32 bytes) -> 2x b128 loads, no packing movs.
__device__ __forceinline__ v16bf load_frag(const unsigned short* p, int lane) {
  return __builtin_bit_cast(v16bf, *reinterpret_cast<const v16us*>(p + lane * 16));
}

__device__ __forceinline__ v8f wmma_bf16(v16bf a, v16bf b, v8f c) {
  return __builtin_amdgcn_wmma_f32_16x16x32_bf16(false, a, false, b, (short)0, c, false, false);
}

// A-fragment element placement (ISA 7.12.2, 16-bit A 16x32):
// lane = half*16 + m ; slot j holds kofs = ((j<8)? j : 8 + j) ... precisely:
//   j in 0..7  -> kofs = j + 8*half
//   j in 8..15 -> kofs = 16 + (j-8) + 8*half
// inverse (kofs -> half,j): half=(kofs>>3)&1; base=kofs-8*half; j=(base<8)?base:base-8;
__device__ __forceinline__ void a_slot(int kofs, int m, int* lane_t, int* j) {
  int half = (kofs >> 3) & 1;
  int base = kofs - 8 * half;
  *j = (base < 8) ? base : base - 8;
  *lane_t = half * 16 + m;
}

// ---------- setup kernels ----------

__global__ void k_pos(const float* __restrict__ frac, const float* __restrict__ lengths,
                      const float* __restrict__ angles, const int* __restrict__ batch,
                      float* __restrict__ pos) {
  int n = blockIdx.x * blockDim.x + threadIdx.x;
  if (n >= NATOM) return;
  int b = batch[n];
  float a  = lengths[b * 3 + 0], bb = lengths[b * 3 + 1], c = lengths[b * 3 + 2];
  const float d2r = 0.017453292519943295f;
  float ca = cosf(angles[b * 3 + 0] * d2r);
  float cb = cosf(angles[b * 3 + 1] * d2r);
  float cg = cosf(angles[b * 3 + 2] * d2r);
  float sg = sinf(angles[b * 3 + 2] * d2r);
  float v3y = (ca - cb * cg) / sg;
  float v3z = sqrtf(fmaxf(1.0f - cb * cb - v3y * v3y, 1e-8f));
  float f0 = frac[n * 3 + 0], f1 = frac[n * 3 + 1], f2 = frac[n * 3 + 2];
  pos[n * 3 + 0] = f0 * a + f1 * bb * cg + f2 * c * cb;
  pos[n * 3 + 1] = f1 * bb * sg + f2 * c * v3y;
  pos[n * 3 + 2] = f2 * c * v3z;
}

__global__ void k_edge(const float* __restrict__ pos, const int* __restrict__ ei,
                       float* __restrict__ dist, float* __restrict__ dirn,
                       float* __restrict__ Yg) {
  int e = blockIdx.x * blockDim.x + threadIdx.x;
  if (e >= NEDGE) return;
  int s = ei[e], d = ei[NEDGE + e];
  float vx = pos[s * 3 + 0] - pos[d * 3 + 0];
  float vy = pos[s * 3 + 1] - pos[d * 3 + 1];
  float vz = pos[s * 3 + 2] - pos[d * 3 + 2];
  float dd = sqrtf(vx * vx + vy * vy + vz * vz) + 1e-8f;
  dist[e] = dd;
  float x = vx / dd, y = vy / dd, ct = vz / dd;
  dirn[e * 3 + 0] = x; dirn[e * 3 + 1] = y; dirn[e * 3 + 2] = ct;

  float rho = sqrtf(x * x + y * y) + 1e-12f;
  float cph = x / rho, sph = y / rho;
  float p0[7], p1[7];
  p0[0] = 1.0f; p0[1] = ct;
#pragma unroll
  for (int l = 2; l <= 6; ++l)
    p0[l] = ((2.0f * l - 1.0f) * ct * p0[l - 1] - (float)(l - 1) * p0[l - 2]) / (float)l;
  p1[0] = 0.0f; p1[1] = -rho; p1[2] = -3.0f * ct * rho;
#pragma unroll
  for (int l = 3; l <= 6; ++l)
    p1[l] = ((2.0f * l - 1.0f) * ct * p1[l - 1] - (float)l * p1[l - 2]) / (float)(l - 1);

  float* Yr = Yg + e * K_SPH;
  Yr[0] = 0.28209479177387814f;  // sqrt(1/4pi)
  int idx = 1;
#pragma unroll
  for (int l = 1; l <= 6; ++l) {
    float K0 = sqrtf((2.0f * l + 1.0f) / FOUR_PI);
    float K1 = sqrtf(2.0f * (2.0f * l + 1.0f) / FOUR_PI / (float)(l * (l + 1)));
    Yr[idx++] = K1 * p1[l] * sph;
    Yr[idx++] = K0 * p0[l];
    Yr[idx++] = K1 * p1[l] * cph;
  }
}

// rbf generated directly in A-fragment order: [tile][kk 0..3][lane][16]
__global__ void k_rbf(const float* __restrict__ dist, unsigned short* __restrict__ rbf_sw) {
  int idx = blockIdx.x * blockDim.x + threadIdx.x;
  if (idx >= NEDGE * NB) return;
  int tile = idx >> 11;          // 2048 elements per 16-edge tile
  int r    = idx & 2047;
  int kk   = r >> 9;             // 512 per fragment
  int lane = (r >> 4) & 31;
  int j    = r & 15;
  int half = lane >> 4, m = lane & 15;
  int i = j >> 1, low = j & 1;
  int kofs = ((i < 4) ? 2 * i : 16 + 2 * (i - 4)) + half * 8 + low;
  int col  = kk * 32 + kofs;     // rbf basis index 0..127
  int e    = tile * 16 + m;
  float width = 8.0f / 127.0f;
  float off = width * (float)col;
  float t = (dist[e] - off) / width;
  rbf_sw[idx] = f2bf(__expf(-0.5f * t * t));
}

__global__ void k_zp(const float* __restrict__ z, const float* __restrict__ zproj,
                     float* __restrict__ zp) {
  int idx = blockIdx.x * blockDim.x + threadIdx.x;
  if (idx >= N_CRYST * CH) return;
  int b = idx >> 7, c = idx & 127;
  float acc = 0.0f;
  for (int j = 0; j < ZDIM; ++j) acc += z[b * ZDIM + j] * zproj[j * CH + c];
  zp[idx] = acc;
}

__global__ void k_xinit(const float* __restrict__ emb, const int* __restrict__ atype,
                        const int* __restrict__ batch, const float* __restrict__ zp,
                        float* __restrict__ x) {
  int idx = blockIdx.x * blockDim.x + threadIdx.x;
  if (idx >= NATOM * KC) return;
  int n = idx / KC, r = idx % KC;
  int k = r >> 7, c = r & 127;
  float v = 0.0f;
  if (k == 0) v = emb[atype[n] * CH + c] + zp[batch[n] * CH + c];
  x[idx] = v;
}

// [W1;Wd] -> B-fragment order: [layer][kk 0..7][nt 0..15][lane][16]
// B layout: lane = half*16 + n_local, slot j holds k = kk*32 + half*8 + ((j<8) ? j : 8+j)
__global__ void k_wcat(const float* __restrict__ W1, const float* __restrict__ Wd,
                       unsigned short* __restrict__ wcat) {
  int idx = blockIdx.x * blockDim.x + threadIdx.x;
  if (idx >= NLAYER * 256 * 256) return;
  int l = idx >> 16, r = idx & 65535;
  int kk   = r >> 13;            // 16 nt * 512
  int nt   = (r >> 9) & 15;
  int lane = (r >> 4) & 31;
  int j    = r & 15;
  int k = kk * 32 + (lane >> 4) * 8 + ((j < 8) ? j : 8 + j);
  int n = nt * 16 + (lane & 15);
  float v = (k < 128) ? W1[l * 128 * 256 + k * 256 + n]
                      : Wd[l * 128 * 256 + (k - 128) * 256 + n];
  wcat[idx] = f2bf(v);
}

// W2 -> B-fragment order: [layer][kk 0..7][nt 0..7][lane][16]
__global__ void k_w2c(const float* __restrict__ W2, unsigned short* __restrict__ w2c) {
  int idx = blockIdx.x * blockDim.x + threadIdx.x;
  if (idx >= NLAYER * 256 * 128) return;
  int l = idx >> 15, r = idx & 32767;
  int kk   = r >> 12;            // 8 nt * 512
  int nt   = (r >> 9) & 7;
  int lane = (r >> 4) & 31;
  int j    = r & 15;
  int k = kk * 32 + (lane >> 4) * 8 + ((j < 8) ? j : 8 + j);
  int n = nt * 16 + (lane & 15);
  w2c[idx] = f2bf(W2[l * 256 * 128 + k * 128 + n]);
}

// ---------- per-layer kernels ----------

// One wave per 16-edge tile; fragments staged in LDS in WMMA layout, b128 loads only.
__global__ __launch_bounds__(128) void k_layer_edge(
    const float* __restrict__ x, const float* __restrict__ Yg,
    const unsigned short* __restrict__ rbf_sw,
    const unsigned short* __restrict__ wcat, const unsigned short* __restrict__ w2c,
    const int* __restrict__ ei, float* __restrict__ mg, int layer) {
  __shared__ float Yl[4][16][K_SPH];
  __shared__ int   srcl[4][16];
  __shared__ alignas(32) unsigned short slf[4][4][32][16];   // s in A-frag order, 16KB
  __shared__ alignas(32) unsigned short hlf[4][8][32][16];   // h in A-frag order, 32KB

  const int w = threadIdx.x >> 5, lane = threadIdx.x & 31;
  const int tile = blockIdx.x * 4 + w;
  const int e0 = tile * 16;

  for (int idx = lane; idx < 16 * K_SPH; idx += 32) {
    int m = idx / K_SPH, k = idx % K_SPH;
    Yl[w][m][k] = Yg[(e0 + m) * K_SPH + k];
  }
  if (lane < 16) srcl[w][lane] = ei[e0 + lane];

  // s[m][c] = sum_k Y[m][k] * x[src_m][k][c]  -> written into A-fragment slots
  for (int idx = lane; idx < 16 * CH; idx += 32) {
    int m = idx >> 7, c = idx & 127;
    const float* xr = x + srcl[w][m] * KC + c;
    float acc = 0.0f;
#pragma unroll
    for (int k = 0; k < K_SPH; ++k) acc += Yl[w][m][k] * xr[k * CH];
    int lt, j; a_slot(c & 31, m, &lt, &j);
    slf[w][c >> 5][lt][j] = f2bf(acc);
  }

  // h = silu([s | rbf] @ Wcat[layer]):  M=16, N=256, K=256.  4 N-tiles per pass.
  const unsigned short* Wl = wcat + layer * (256 * 256);
  __builtin_prefetch(Wl + lane * 64, 0, 0);
  for (int ntg = 0; ntg < 4; ++ntg) {
    v8f acc[4] = {{}, {}, {}, {}};
#pragma unroll
    for (int kk = 0; kk < 8; ++kk) {
      v16bf A = (kk < 4) ? load_frag(&slf[w][kk][0][0], lane)
                         : load_frag(rbf_sw + tile * 2048 + (kk - 4) * 512, lane);
#pragma unroll
      for (int q = 0; q < 4; ++q) {
        int nt = ntg * 4 + q;
        v16bf B = load_frag(Wl + (kk * 16 + nt) * 512, lane);
        acc[q] = wmma_bf16(A, B, acc[q]);
      }
    }
    int n = lane & 15, hh = lane >> 4;
#pragma unroll
    for (int q = 0; q < 4; ++q) {
      int col0 = (ntg * 4 + q) * 16 + n;
#pragma unroll
      for (int r = 0; r < 8; ++r) {
        float v = acc[q][r];
        v = v / (1.0f + __expf(-v));          // silu
        int lt, j; a_slot(col0 & 31, r + 8 * hh, &lt, &j);
        hlf[w][col0 >> 5][lt][j] = f2bf(v);
      }
    }
  }

  // m = h @ W2[layer]:  M=16, N=128, K=256.  All 8 N-tiles live in registers.
  const unsigned short* W2l = w2c + layer * (256 * 128);
  __builtin_prefetch(W2l + lane * 64, 0, 0);
  {
    v8f acc[8] = {{}, {}, {}, {}, {}, {}, {}, {}};
#pragma unroll
    for (int kk = 0; kk < 8; ++kk) {
      v16bf A = load_frag(&hlf[w][kk][0][0], lane);
#pragma unroll
      for (int nt = 0; nt < 8; ++nt) {
        v16bf B = load_frag(W2l + (kk * 8 + nt) * 512, lane);
        acc[nt] = wmma_bf16(A, B, acc[nt]);
      }
    }
    int n = lane & 15, hh = lane >> 4;
#pragma unroll
    for (int nt = 0; nt < 8; ++nt)
#pragma unroll
      for (int r = 0; r < 8; ++r)
        mg[(e0 + r + 8 * hh) * CH + nt * 16 + n] = acc[nt][r];
  }
}

// dst = e/20 is contiguous: per-atom reduction over its 20 edges, no atomics.
__global__ __launch_bounds__(128) void k_aggregate(const float* __restrict__ mg,
                                                   const float* __restrict__ Yg,
                                                   float* __restrict__ x) {
  int n = blockIdx.x, c = threadIdx.x;
  __shared__ float Yb[NBRS][K_SPH];
  for (int idx = threadIdx.x; idx < NBRS * K_SPH; idx += 128)
    Yb[idx / K_SPH][idx % K_SPH] = Yg[(n * NBRS + idx / K_SPH) * K_SPH + idx % K_SPH];
  __syncthreads();
  float mv[NBRS];
#pragma unroll
  for (int j = 0; j < NBRS; ++j) mv[j] = mg[(n * NBRS + j) * CH + c];
  float* xp = x + n * KC + c;
#pragma unroll
  for (int k = 0; k < K_SPH; ++k) {
    float a = 0.0f;
#pragma unroll
    for (int j = 0; j < NBRS; ++j) a += Yb[j][k] * mv[j];
    xp[k * CH] += a * (1.0f / (float)NBRS);
  }
}

// ---------- readout ----------

__global__ __launch_bounds__(128) void k_logits(const float* __restrict__ x,
                                                const float* __restrict__ Wa,
                                                const float* __restrict__ ba,
                                                float* __restrict__ out) {
  int n = blockIdx.x, t = threadIdx.x;
  __shared__ float hn[CH];
  hn[t] = x[n * KC + t];
  __syncthreads();
  if (t < MAXZ) {
    float a = ba[t];
    for (int k = 0; k < CH; ++k) a += hn[k] * Wa[k * MAXZ + t];
    out[NATOM * 3 + n * MAXZ + t] = a;
  }
}

__global__ __launch_bounds__(128) void k_fscal(const float* __restrict__ x,
                                               const float* __restrict__ wf,
                                               const int* __restrict__ ei,
                                               float* __restrict__ fscal) {
  int e = blockIdx.x * 4 + (threadIdx.x >> 5);
  int lane = threadIdx.x & 31;
  int s = ei[e], d = ei[NEDGE + e];
  float a = 0.0f;
  for (int c = lane; c < CH; c += 32)
    a += (x[s * KC + c] - x[d * KC + c]) * wf[c];
#pragma unroll
  for (int off = 16; off > 0; off >>= 1) a += __shfl_xor(a, off, 32);
  if (lane == 0) fscal[e] = a;
}

__global__ void k_coord(const float* __restrict__ fscal, const float* __restrict__ dirn,
                        float* __restrict__ out) {
  int n = blockIdx.x * blockDim.x + threadIdx.x;
  if (n >= NATOM) return;
  float a0 = 0.0f, a1 = 0.0f, a2 = 0.0f;
#pragma unroll
  for (int j = 0; j < NBRS; ++j) {
    int e = n * NBRS + j;
    float f = fscal[e];
    a0 += f * dirn[e * 3 + 0];
    a1 += f * dirn[e * 3 + 1];
    a2 += f * dirn[e * 3 + 2];
  }
  out[n * 3 + 0] = a0; out[n * 3 + 1] = a1; out[n * 3 + 2] = a2;
}

// ---------- host ----------

extern "C" void kernel_launch(void* const* d_in, const int* in_sizes, int n_in,
                              void* d_out, int out_size, void* d_ws, size_t ws_size,
                              hipStream_t stream) {
  (void)in_sizes; (void)n_in; (void)out_size; (void)ws_size;
  const float* z      = (const float*)d_in[0];
  const float* frac   = (const float*)d_in[1];
  const int*   atype  = (const int*)d_in[2];
  const float* lengths= (const float*)d_in[4];
  const float* angles = (const float*)d_in[5];
  const int*   batch  = (const int*)d_in[6];
  const int*   ei     = (const int*)d_in[7];
  const float* emb    = (const float*)d_in[8];
  const float* zproj  = (const float*)d_in[9];
  const float* Wd     = (const float*)d_in[10];
  const float* W1     = (const float*)d_in[11];
  const float* W2     = (const float*)d_in[12];
  const float* Wa     = (const float*)d_in[13];
  const float* ba     = (const float*)d_in[14];
  const float* wf     = (const float*)d_in[15];
  float* out = (float*)d_out;

  float* pos   = (float*)d_ws;
  float* dist  = pos + NATOM * 3;
  float* dirn  = dist + NEDGE;
  float* Yg    = dirn + NEDGE * 3;
  float* zp    = Yg + NEDGE * K_SPH;
  float* x     = zp + N_CRYST * CH;
  float* mg    = x + NATOM * KC;
  float* fscal = mg + NEDGE * CH;
  unsigned short* rbf_sw = (unsigned short*)(fscal + NEDGE);
  unsigned short* wcat   = rbf_sw + NEDGE * NB;
  unsigned short* w2c    = wcat + NLAYER * 256 * 256;

  k_pos  <<<(NATOM + 127) / 128, 128, 0, stream>>>(frac, lengths, angles, batch, pos);
  k_edge <<<(NEDGE + 127) / 128, 128, 0, stream>>>(pos, ei, dist, dirn, Yg);
  k_rbf  <<<(NEDGE * NB) / 256, 256, 0, stream>>>(dist, rbf_sw);
  k_zp   <<<(N_CRYST * CH) / 128, 128, 0, stream>>>(z, zproj, zp);
  k_xinit<<<(NATOM * KC + 255) / 256, 256, 0, stream>>>(emb, atype, batch, zp, x);
  k_wcat <<<(NLAYER * 256 * 256) / 256, 256, 0, stream>>>(W1, Wd, wcat);
  k_w2c  <<<(NLAYER * 256 * 128) / 256, 256, 0, stream>>>(W2, w2c);

  for (int l = 0; l < NLAYER; ++l) {
    k_layer_edge<<<NEDGE / 64, 128, 0, stream>>>(x, Yg, rbf_sw, wcat, w2c, ei, mg, l);
    k_aggregate <<<NATOM, 128, 0, stream>>>(mg, Yg, x);
  }

  k_logits<<<NATOM, 128, 0, stream>>>(x, Wa, ba, out);
  k_fscal <<<NEDGE / 4, 128, 0, stream>>>(x, wf, ei, fscal);
  k_coord <<<(NATOM + 127) / 128, 128, 0, stream>>>(fscal, dirn, out);
}
